// Inducer_28870770164393
// MI455X (gfx1250) — matrix operand order repaired and checked by reading, not compiled, not run on hardware
//
#include <hip/hip_runtime.h>
#include <hip/hip_bf16.h>
#include <cstdint>

typedef __attribute__((ext_vector_type(16))) _Float16 v16h;
typedef __attribute__((ext_vector_type(8)))  float    v8f;

#define KSAMP 4096
#define NTOK  20
#define DVEC  256
#define NSTEP 19
// gumbel table: (19,4096,20) -> 1,556,480 elems, threefry half = 778,240
#define GUM_N 1556480u
#define GUM_H 778240u
// op table: (19,4096) -> 77,824 elems, half = 38,912
#define OPS_H 38912u

#ifndef __has_builtin
#define __has_builtin(x) 0
#endif
#if __has_builtin(__builtin_amdgcn_global_load_async_to_lds_b128) && \
    __has_builtin(__builtin_amdgcn_s_wait_asynccnt)
#define USE_ASYNC_LDS 1
#else
#define USE_ASYNC_LDS 0
#endif

#if USE_ASYNC_LDS
typedef int v4i __attribute__((vector_size(16)));
typedef __attribute__((address_space(1))) v4i gas_v4i;  // global
typedef __attribute__((address_space(3))) v4i las_v4i;  // LDS
#endif

// ---------------------------------------------------------------------------
// Threefry-2x32-20 (JAX PRNG core)
// ---------------------------------------------------------------------------
__device__ __forceinline__ uint32_t rotl32(uint32_t x, int r) {
  return (x << r) | (x >> (32 - r));
}

__device__ __forceinline__ void tf2x32(uint32_t k0, uint32_t k1,
                                       uint32_t c0, uint32_t c1,
                                       uint32_t& o0, uint32_t& o1) {
  uint32_t ks2 = 0x1BD11BDAu ^ k0 ^ k1;
  uint32_t x0 = c0 + k0, x1 = c1 + k1;
  const int R0[4] = {13, 15, 26, 6};
  const int R1[4] = {17, 29, 16, 24};
#pragma unroll
  for (int i = 0; i < 4; ++i) { x0 += x1; x1 = rotl32(x1, R0[i]); x1 ^= x0; }
  x0 += k1;  x1 += ks2 + 1u;
#pragma unroll
  for (int i = 0; i < 4; ++i) { x0 += x1; x1 = rotl32(x1, R1[i]); x1 ^= x0; }
  x0 += ks2; x1 += k0 + 2u;
#pragma unroll
  for (int i = 0; i < 4; ++i) { x0 += x1; x1 = rotl32(x1, R0[i]); x1 ^= x0; }
  x0 += k0;  x1 += k1 + 3u;
#pragma unroll
  for (int i = 0; i < 4; ++i) { x0 += x1; x1 = rotl32(x1, R1[i]); x1 ^= x0; }
  x0 += k1;  x1 += ks2 + 4u;
#pragma unroll
  for (int i = 0; i < 4; ++i) { x0 += x1; x1 = rotl32(x1, R0[i]); x1 ^= x0; }
  o0 = x0 + ks2;
  o1 = x1 + k0 + 5u;
}

__device__ __forceinline__ float gumbel_of(uint32_t bits) {
  float u = __uint_as_float((bits >> 9) | 0x3F800000u) - 1.0f;
  u = fmaxf(u, 1.17549435e-38f);
  return -logf(-logf(u));
}

// derive kop/kg = split(key(42)) once (counts {0,1,2,3}, halves-split layout)
__device__ __forceinline__ void derive_keys(uint32_t& kop0, uint32_t& kop1,
                                            uint32_t& kg0, uint32_t& kg1) {
  uint32_t a0, a1, b0, b1;
  tf2x32(0u, 42u, 0u, 2u, a0, a1);
  tf2x32(0u, 42u, 1u, 3u, b0, b1);
  kop0 = a0; kop1 = b0; kg0 = a1; kg1 = b1;
}

// ---------------------------------------------------------------------------
// Kernel 1: x = softmax(emb[ids]) * learn[ids] + fixed[ids]; emit f16-padded
// A (32x256) and B (256x16 = [W_func | W_arg | 0]) operands for WMMA.
// ---------------------------------------------------------------------------
__global__ void k_build_x(const int* __restrict__ ids,
                          const float* __restrict__ emb,
                          const float* __restrict__ learn,
                          const float* __restrict__ fixedv,
                          const float* __restrict__ Wf,
                          const float* __restrict__ Wa,
                          _Float16* __restrict__ xh,
                          _Float16* __restrict__ wh) {
  const int r = blockIdx.x;   // 0..31 (rows 20..31 are zero padding)
  const int d = threadIdx.x;  // 0..255
  __shared__ float red[256];

  if (r >= NTOK) {
    xh[r * DVEC + d] = (_Float16)0.0f;
  } else {
    const int id = ids[r];
    const float e = emb[(size_t)id * DVEC + d];
    red[d] = e;
    __syncthreads();
#pragma unroll
    for (int s = 128; s > 0; s >>= 1) {
      if (d < s) red[d] = fmaxf(red[d], red[d + s]);
      __syncthreads();
    }
    const float mx = red[0];
    __syncthreads();
    const float ex = expf(e - mx);
    red[d] = ex;
    __syncthreads();
#pragma unroll
    for (int s = 128; s > 0; s >>= 1) {
      if (d < s) red[d] += red[d + s];
      __syncthreads();
    }
    const float sum = red[0];
    const float xv = (ex / sum) * learn[id] + fixedv[(size_t)id * DVEC + d];
    xh[r * DVEC + d] = (_Float16)xv;
  }

  if (r == 0) {
#pragma unroll
    for (int j = 0; j < 16; ++j) {
      float w = (j < 3) ? Wf[d * 3 + j] : ((j < 6) ? Wa[d * 3 + (j - 3)] : 0.0f);
      wh[d * 16 + j] = (_Float16)w;
    }
  }
}

// ---------------------------------------------------------------------------
// Kernel 2: P(32x16) = A(32x256) @ B(256x16) with v_wmma_f32_16x16x32_f16.
// ---------------------------------------------------------------------------
__global__ void k_wmma_proj(const _Float16* __restrict__ xh,
                            const _Float16* __restrict__ wh,
                            float* __restrict__ P) {
  const int lane = threadIdx.x;  // 0..31, EXEC all ones
  const int lrow = lane & 15;
  const bool hi  = (lane >= 16);

  for (int m = 0; m < 2; ++m) {
    v8f acc = {};
#pragma unroll
    for (int kc = 0; kc < 8; ++kc) {
      v16h a, b;
      const int M = m * 16 + lrow;
      const int ka = hi ? 8 : 0;
#pragma unroll
      for (int h = 0; h < 16; ++h) {
        const int v = h >> 1, e = h & 1;
        const int K = ((v < 4) ? 0 : 16) + 2 * (v & 3) + e + ka + kc * 32;
        a[h] = xh[M * DVEC + K];
      }
      const int kb = hi ? 8 : 0;
#pragma unroll
      for (int h = 0; h < 16; ++h) {
        const int K = ((h < 8) ? h : h + 8) + kb + kc * 32;
        b[h] = wh[K * 16 + lrow];
      }
      acc = __builtin_amdgcn_wmma_f32_16x16x32_f16(
          false, a, false, b, (short)0, acc, false, false);
    }
    const int mbase = m * 16 + (hi ? 8 : 0);
#pragma unroll
    for (int r = 0; r < 8; ++r) P[(mbase + r) * 16 + lrow] = acc[r];
  }
}

// ---------------------------------------------------------------------------
// Kernel G: massively parallel PRNG. One threefry pair per thread -> two
// gumbels (elements i and i+H of the flat (19,4096,20) array). First 38912
// threads also produce the op pairs for the (19,4096) randint array.
// Tables total ~6.3 MB: L2-resident on MI455X (192 MB L2).
// ---------------------------------------------------------------------------
__global__ void k_genrand(float* __restrict__ gum,
                          unsigned char* __restrict__ ops) {
  const uint32_t i = blockIdx.x * blockDim.x + threadIdx.x;  // 0..778239
  uint32_t kop0, kop1, kg0, kg1;
  derive_keys(kop0, kop1, kg0, kg1);

  uint32_t o0, o1;
  tf2x32(kg0, kg1, i, i + GUM_H, o0, o1);
  gum[i]         = gumbel_of(o0);
  gum[i + GUM_H] = gumbel_of(o1);

  if (i < OPS_H) {
    uint32_t p0, p1;
    tf2x32(kop0, kop1, i, i + OPS_H, p0, p1);
    ops[i]         = (unsigned char)(((p0 >> 16) + (p0 & 0xFFFFu)) % 3u);
    ops[i + OPS_H] = (unsigned char)(((p1 >> 16) + (p1 & 0xFFFFu)) % 3u);
  }
}

// ---------------------------------------------------------------------------
// Kernel 3: Monte-Carlo scan. One thread per sample; per-step gumbel slab
// (256 samples x 20 floats = 20KB) double-buffered in LDS, prefetched with
// async global->LDS loads (when the gfx1250 builtins are available).
// ---------------------------------------------------------------------------
#define SLAB (256 * NTOK)  // floats per step per block

__global__ void __launch_bounds__(256)
k_mc_scan(const float* __restrict__ P,
          const float* __restrict__ bop,
          const float* __restrict__ gum,
          const unsigned char* __restrict__ ops,
          float* __restrict__ scores_out,
          int* __restrict__ legal_out) {
  __shared__ float Sf[NTOK * 3], Sa[NTOK * 3], Bop[3];
  __shared__ float gbuf[2][SLAB];  // 2 x 20 KB

  const int tid = threadIdx.x;
  const uint32_t k0 = blockIdx.x * 256u;
  const uint32_t k = k0 + (uint32_t)tid;

  if (tid < NTOK * 3) {
    const int i = tid / 3, j = tid % 3;
    Sf[tid] = P[i * 16 + j];
    Sa[tid] = P[i * 16 + 3 + j];
  }
  if (tid < 3) Bop[tid] = bop[tid];

  // ---- stage helper: copy step-t slab into gbuf[buf] ----
  auto stage = [&](int buf, int t) {
    const float* src = gum + ((size_t)t * KSAMP + k0) * NTOK;
#if USE_ASYNC_LDS
    const char* gsrc = (const char*)src;
    char* ldst = (char*)&gbuf[buf][0];
#pragma unroll
    for (int c = 0; c < 5; ++c) {
      const int off = tid * 16 + c * 4096;  // 256 thr x 16B x 5 = 20 KB
      __builtin_amdgcn_global_load_async_to_lds_b128(
          (gas_v4i*)(gsrc + off), (las_v4i*)(ldst + off), 0, 0);
    }
#else
#pragma unroll
    for (int c = 0; c < NTOK; ++c)
      gbuf[buf][tid + c * 256] = src[tid + c * 256];
#endif
  };
  auto stage_wait = [&]() {
#if USE_ASYNC_LDS
    __builtin_amdgcn_s_wait_asynccnt(0);
#endif
    __syncthreads();
  };

  // prime buffer 0 with step 0
  stage(0, 0);
  stage_wait();

  uint32_t avail = (1u << NTOK) - 1u;
  uint32_t f1m = 0u, f2m = 0u;
  uint32_t legal = 1u;
  float score = 0.0f;

  // ptr[slot] in [0,20): 5 bits each; slots 0..11 in plo, 12..19 in phi
  unsigned long long plo = 0ull, phi = 0ull;
#pragma unroll
  for (int s = 0; s < 12; ++s) plo |= (unsigned long long)s << (5 * s);
#pragma unroll
  for (int s = 12; s < NTOK; ++s) phi |= (unsigned long long)s << (5 * (s - 12));

  auto getp = [&](int s) -> uint32_t {
    return (s < 12) ? (uint32_t)((plo >> (5 * s)) & 31ull)
                    : (uint32_t)((phi >> (5 * (s - 12))) & 31ull);
  };
  auto setp = [&](int s, uint32_t v) {
    if (s < 12) {
      const int t = 5 * s;
      plo = (plo & ~(31ull << t)) | ((unsigned long long)v << t);
    } else {
      const int t = 5 * (s - 12);
      phi = (phi & ~(31ull << t)) | ((unsigned long long)v << t);
    }
  };

  for (int t = 0; t < NSTEP; ++t) {
    const int cur = t & 1;
    if (t + 1 < NSTEP) stage(cur ^ 1, t + 1);  // prefetch next slab

    const uint32_t op = (uint32_t)ops[(uint32_t)t * KSAMP + k];

    // masked top-2 over this sample's 20 gumbels (strict > == top_k ties)
    float best = -__builtin_inff(), second = -__builtin_inff();
    int bi = 0, si = 0;
#pragma unroll
    for (int j = 0; j < NTOK; ++j) {
      const float g = gbuf[cur][tid * NTOK + j];
      const float sel = ((avail >> j) & 1u) ? g : -1.0e9f;
      if (sel > best) { second = best; si = bi; best = sel; bi = j; }
      else if (sel > second) { second = sel; si = j; }
    }
    const int ifn = bi, ia = si;

    // score += S_f[ptr[ifn]][op] + S_a[ptr[ia]][op] + b_op[op]
    const uint32_t pf = getp(ifn), pa = getp(ia);
    score += Sf[pf * 3u + op] + Sa[pa * 3u + op] + Bop[op];

    // legality (func-slot flags, pre-update)
    const uint32_t ff1 = (f1m >> ifn) & 1u;
    const uint32_t ff2 = (f2m >> ifn) & 1u;
    const uint32_t bad = (op == 0u) ? ff1 : ((op == 1u) ? (ff1 | ff2) : 0u);
    legal &= (bad ^ 1u);

    // state update
    avail &= ~(1u << ia);
    if (op == 2u) {  // mod: func slot <- full copy of arg slot
      setp(ifn, pa);
      const uint32_t fa1 = (f1m >> ia) & 1u;
      const uint32_t fa2 = (f2m >> ia) & 1u;
      f1m = (f1m & ~(1u << ifn)) | (fa1 << ifn);
      f2m = (f2m & ~(1u << ifn)) | (fa2 << ifn);
    } else if (op == 0u) {
      f1m |= 1u << ifn;
    } else {
      f2m |= 1u << ifn;
    }

    stage_wait();  // next slab resident + all reads of cur slab done
  }

  scores_out[k] = score;
  legal_out[k] = (int)legal;
}

// ---------------------------------------------------------------------------
// Kernel 4: masked logsumexp + legal count -> scalar output
// ---------------------------------------------------------------------------
__global__ void k_reduce(const float* __restrict__ scores,
                         const int* __restrict__ legal,
                         float* __restrict__ out) {
  __shared__ float smax[256];
  __shared__ float ssum[256];
  __shared__ int scnt[256];
  const int tid = threadIdx.x;

  float mx = -__builtin_inff();
  for (int i = tid; i < KSAMP; i += 256)
    if (legal[i]) mx = fmaxf(mx, scores[i]);
  smax[tid] = mx;
  __syncthreads();
#pragma unroll
  for (int s = 128; s > 0; s >>= 1) {
    if (tid < s) smax[tid] = fmaxf(smax[tid], smax[tid + s]);
    __syncthreads();
  }
  const float M = smax[0];

  float sum = 0.0f;
  int cnt = 0;
  for (int i = tid; i < KSAMP; i += 256)
    if (legal[i]) { sum += expf(scores[i] - M); cnt++; }
  ssum[tid] = sum;
  scnt[tid] = cnt;
  __syncthreads();
#pragma unroll
  for (int s = 128; s > 0; s >>= 1) {
    if (tid < s) { ssum[tid] += ssum[tid + s]; scnt[tid] += scnt[tid + s]; }
    __syncthreads();
  }

  if (tid == 0) {
    // log(20!) + log(Catalan[19]) + 19*log(3) - 2*log(4096)
    const double log_all_trees = log(2432902008176640000.0) +
                                 log(1767263190.0) + 19.0 * log(3.0);
    const double res = log_all_trees + log((double)scnt[0]) -
                       2.0 * log(4096.0) + (double)M + log((double)ssum[0]);
    out[0] = (float)res;
  }
}

// ---------------------------------------------------------------------------
// Launch
// ---------------------------------------------------------------------------
extern "C" void kernel_launch(void* const* d_in, const int* in_sizes, int n_in,
                              void* d_out, int out_size, void* d_ws, size_t ws_size,
                              hipStream_t stream) {
  (void)in_sizes; (void)n_in; (void)out_size; (void)ws_size;
  const int*   ids    = (const int*)d_in[0];
  const float* emb    = (const float*)d_in[1];
  const float* learn  = (const float*)d_in[2];
  const float* fixedv = (const float*)d_in[3];
  const float* Wf     = (const float*)d_in[4];
  const float* Wa     = (const float*)d_in[5];
  const float* bop    = (const float*)d_in[6];
  float* out = (float*)d_out;

  char* ws = (char*)d_ws;
  _Float16*      xh  = (_Float16*)(ws);              // 16384 B
  _Float16*      wh  = (_Float16*)(ws + 16384);      //  8192 B
  float*         P   = (float*)(ws + 24576);         //  2048 B
  float*         sc  = (float*)(ws + 26624);         // 16384 B
  int*           lg  = (int*)(ws + 43008);           // 16384 B
  float*         gum = (float*)(ws + 59392);         // 6,225,920 B
  unsigned char* opb = (unsigned char*)(ws + 59392 + 6225920);  // 77,824 B

  k_build_x<<<32, 256, 0, stream>>>(ids, emb, learn, fixedv, Wf, Wa, xh, wh);
  k_wmma_proj<<<1, 32, 0, stream>>>(xh, wh, P);
  k_genrand<<<GUM_H / 256, 256, 0, stream>>>(gum, opb);
  k_mc_scan<<<KSAMP / 256, 256, 0, stream>>>(P, bop, gum, opb, sc, lg);
  k_reduce<<<1, 256, 0, stream>>>(sc, lg, out);
}
